// DenoiseFlow_42872363548709
// MI455X (gfx1250) — compile-verified, matrix operand order checked by reading, not run on hardware
//
#include <hip/hip_runtime.h>
#include <hip/hip_bf16.h>
#include <utility>

#define B_SZ   8
#define N_PTS  1024
#define PC     3
#define AUG    24
#define IDIM   27
#define C1     14
#define C2     13
#define HDIM   128
#define NFLOW  12
#define A_HID  64
#define A_C1   12
#define AIN    (A_C1 + A_HID)   /* 76 */
#define KMAX   24
#define CUT    6
#define NU     (NFLOW * 4)
#define NODES  4                /* nodes (waves) per edgeconv block */
#define LOG2PI 1.8378770664093453f

typedef __attribute__((ext_vector_type(16))) _Float16 v16h;
typedef __attribute__((ext_vector_type(8)))  _Float16 v8h;
typedef __attribute__((ext_vector_type(8)))  float    v8f;

// ---- Tensor Data Mover availability / arity probes ----
#if defined(__has_builtin)
#  if __has_builtin(__builtin_amdgcn_tensor_load_to_lds) && \
      __has_builtin(__builtin_amdgcn_s_wait_tensorcnt)
#    define HAVE_TDM 1
#  endif
#endif
#if defined(__clang_major__) && (__clang_major__ >= 23)
#  define TDM_6ARG 1
#endif

typedef __attribute__((ext_vector_type(4))) unsigned int u32x4;
typedef __attribute__((ext_vector_type(8))) int          i32x8;
typedef __attribute__((ext_vector_type(4))) int          i32x4;

// ---------------------------------------------------------------------------
// WMMA fragment mapping (ISA 7.12.2, wave32):
// A (16xK): lane l<16 row l, elements = K {k0..k0+7, k0+16..k0+23};
//           lane l+16 same row, K {k0+8..15, k0+24..31}. Two contiguous 8-half
//           runs -> 2x ds_load_b128.
// B (Kx16): col = lane&15, same K mapping -> pre-swizzled contiguous 32B/lane.
// C/D     : col = lane&15, row = vgpr + 8*(lane>=16).
// ---------------------------------------------------------------------------
__device__ __forceinline__ int kmap_e(int hi, int e) {
  int k = (e < 8) ? e : (e + 8);
  return k + (hi ? 8 : 0);
}

__device__ __forceinline__ v16h load_a_frag(const _Float16* sm, int lane, int k0, int ld) {
  int row = lane & 15, hi = lane >> 4;
  const _Float16* p = sm + row * ld + k0 + hi * 8;
  v8h lo = *(const v8h*)p;
  v8h hh = *(const v8h*)(p + 16);
  return __builtin_shufflevector(lo, hh, 0, 1, 2, 3, 4, 5, 6, 7, 8, 9, 10, 11, 12, 13, 14, 15);
}

__device__ __forceinline__ v8f wmma_f16(v16h a, v16h b, v8f c) {
  return __builtin_amdgcn_wmma_f32_16x16x32_f16(false, a, false, b, (short)0, c, false, false);
}

// ---------------------------------------------------------------------------
// TDM 1-D copy global->LDS.  D# per ISA ch.8: group0 {count, lds_addr,
// global_addr, type=2}; group1 {data_size=8B, dim0=tile0=bytes/8, dim1=1,
// stride0=bytes/8}; groups 2/3 zero (<=2D tensor).  Tracked by TENSORcnt.
// ---------------------------------------------------------------------------
#if defined(HAVE_TDM)
__device__ __forceinline__ void tdm_copy_1d(void* lds_dst, const void* gsrc, unsigned bytes) {
  unsigned n8 = bytes >> 3;  // 8-byte units
  unsigned lds_addr = (unsigned)(size_t)lds_dst;       // flat low 32 = LDS byte offset
  unsigned long long ga = (unsigned long long)(size_t)gsrc;
  u32x4 g0;
  g0[0] = 1u;                                          // count=1, user descriptor
  g0[1] = lds_addr;                                    // lds_addr  [63:32]
  g0[2] = (unsigned)(ga & 0xffffffffu);                // global_addr lo
  g0[3] = (unsigned)((ga >> 32) & 0x01ffffffu) | 0x80000000u;  // addr[56:32] | type=2
  i32x8 g1;
  g1[0] = (int)(3u << 16);                             // data_size = 8 bytes
  g1[1] = (int)((n8 & 0xffffu) << 16);                 // tensor_dim0[15:0]
  g1[2] = (int)((n8 >> 16) & 0xffffu) | (1 << 16);     // tensor_dim0[31:16] | tensor_dim1=1
  g1[3] = (int)((n8 & 0xffffu) << 16);                 // tile_dim0
  g1[4] = 1;                                           // tile_dim1 = 1
  g1[5] = (int)n8;                                     // tensor_dim0_stride lo32
  g1[6] = 0;
  g1[7] = 0;
  i32x4 z4 = {0, 0, 0, 0};
#if defined(TDM_6ARG)
  i32x8 z8 = {0, 0, 0, 0, 0, 0, 0, 0};
  __builtin_amdgcn_tensor_load_to_lds(g0, g1, z4, z4, z8, 0);
#else
  __builtin_amdgcn_tensor_load_to_lds(g0, g1, z4, z4, 0);
#endif
}
#endif

// ---------------------------------------------------------------------------
// Weight prep: f32 -> f16, pre-swizzled into WMMA B-fragment order.
// W1s: [u][ct=8][lane=32][e=16]   (K 28 padded to 32)
// W2s: [u][ct=8][chunk=4][lane=32][e=16]
// ---------------------------------------------------------------------------
__global__ void prep_w1_kernel(const float* kW1, _Float16* W1s) {
  int t = blockIdx.x * 256 + threadIdx.x;
  if (t >= NU * 4096) return;
  int e = t & 15, lane = (t >> 4) & 31, ct = (t >> 9) & 7, u = t >> 12;
  int k = kmap_e(lane >> 4, e);
  int col = ct * 16 + (lane & 15);
  float v = (k < 2 * C1) ? kW1[((size_t)u * (2 * C1) + k) * HDIM + col] : 0.f;
  W1s[t] = (_Float16)v;
}

__global__ void prep_w2_kernel(const float* kW2, _Float16* W2s) {
  int t = blockIdx.x * 256 + threadIdx.x;
  if (t >= NU * 16384) return;
  int e = t & 15, lane = (t >> 4) & 31, chunk = (t >> 9) & 3, ct = (t >> 11) & 7, u = t >> 14;
  int k = chunk * 32 + kmap_e(lane >> 4, e);
  int col = ct * 16 + (lane & 15);
  W2s[t] = (_Float16)kW2[((size_t)u * HDIM + k) * HDIM + col];
}

// ---------------------------------------------------------------------------
// Brute-force kNN, templated so top-k stays in registers (unrolled bubble)
// ---------------------------------------------------------------------------
template <int K, int C>
__global__ __launch_bounds__(N_PTS)
void knn_kernel_t(const float* pts, int stride, int* idx) {
  __shared__ float sp[N_PTS * C];
  int b = blockIdx.x, n = threadIdx.x;
  const float* base = pts + (size_t)b * N_PTS * stride;
#pragma unroll
  for (int c = 0; c < C; ++c) sp[n * C + c] = base[(size_t)n * stride + c];
  __syncthreads();

  float pn[C];
#pragma unroll
  for (int c = 0; c < C; ++c) pn[c] = sp[n * C + c];

  float bd[K];
  int   bi[K];
#pragma unroll
  for (int i = 0; i < K; ++i) { bd[i] = 1e30f; bi[i] = 0; }

  for (int m = 0; m < N_PTS; ++m) {
    float d = 0.f;
#pragma unroll
    for (int c = 0; c < C; ++c) { float df = pn[c] - sp[m * C + c]; d += df * df; }
    if (d < bd[K - 1]) {
      bd[K - 1] = d; bi[K - 1] = m;
#pragma unroll
      for (int p = K - 1; p > 0; --p) {
        if (bd[p] < bd[p - 1]) {
          float td = bd[p]; bd[p] = bd[p - 1]; bd[p - 1] = td;
          int   ti = bi[p]; bi[p] = bi[p - 1]; bi[p - 1] = ti;
        }
      }
    }
  }
  int* orow = idx + ((size_t)b * N_PTS + n) * KMAX;
#pragma unroll
  for (int i = 0; i < K; ++i) orow[i] = bi[i];
}

// ---------------------------------------------------------------------------
// Fused edge-conv coupling: 128 threads = 4 wave32, one node per wave.
// Weights DMA-staged to LDS via TDM (fallback: cooperative copy);
// everything after staging is wave-local.
// ---------------------------------------------------------------------------
__global__ __launch_bounds__(128)
void edgeconv_kernel(float* x, const int* idx, int kcur,
                     const _Float16* W1s, const _Float16* W2s,
                     const float* kb1, const float* kb2,
                     const float* kW3, const float* kb3,
                     int uS, int uT, int invmode, float* node_ldj) {
  __shared__ __align__(32) _Float16 sEA[NODES][16 * 32];     //  4 KB
  __shared__ __align__(32) _Float16 sH1[NODES][16 * HDIM];   // 16 KB
  __shared__ __align__(32) _Float16 sW1[8 * 32 * 16];        //  8 KB
  __shared__ __align__(32) _Float16 sW2[8 * 4 * 32 * 16];    // 32 KB

  int tid = threadIdx.x, lane = tid & 31, wave = tid >> 5;
  int bn = blockIdx.x * NODES + wave;
  int b = bn >> 10;
  float* xrow = x + (size_t)bn * IDIM;

  // ---- kick off S-net weight staging (overlaps with edge-feature build) ----
#if defined(HAVE_TDM)
  if (wave == 0) {
    tdm_copy_1d(sW1, W1s + (size_t)uS * 4096, 8192);
    tdm_copy_1d(sW2, W2s + (size_t)uS * 16384, 32768);
  }
#else
  {
    const int4* g1 = (const int4*)(W1s + (size_t)uS * 4096);
    const int4* g2 = (const int4*)(W2s + (size_t)uS * 16384);
    int4* s1 = (int4*)sW1;
    int4* s2 = (int4*)sW2;
    for (int i = tid; i < 512; i += 128) s1[i] = g1[i];
    for (int i = tid; i < 2048; i += 128) s2[i] = g2[i];
  }
#endif

  // ---- per-wave edge features e = [xa, nb - xa] (16 rows x 32 cols) ----
  int myidx = 0;
  if (lane < 16) {
    int kc = lane < kcur ? lane : kcur - 1;
    myidx = idx[(size_t)bn * KMAX + kc];
  }
  {
    int c = lane;
    int ch = (c < C1) ? c : c - C1;
    float xc = (c < 2 * C1) ? xrow[ch] : 0.f;
    for (int it = 0; it < 16; ++it) {
      int nbr = __shfl(myidx, it, 32);
      float v = 0.f;
      if (c < 2 * C1)
        v = (c < C1) ? xc : x[((size_t)b * N_PTS + nbr) * IDIM + ch] - xc;
      sEA[wave][it * 32 + c] = (_Float16)v;
    }
  }

  float outv[2][C2];

#pragma unroll 2
  for (int net = 0; net < 2; ++net) {
    int u = net ? uT : uS;
    if (net == 1) {
      __syncthreads();  // all waves done reading S-net weights
#if defined(HAVE_TDM)
      if (wave == 0) {
        tdm_copy_1d(sW1, W1s + (size_t)uT * 4096, 8192);
        tdm_copy_1d(sW2, W2s + (size_t)uT * 16384, 32768);
      }
#else
      {
        const int4* g1 = (const int4*)(W1s + (size_t)uT * 4096);
        const int4* g2 = (const int4*)(W2s + (size_t)uT * 16384);
        int4* s1 = (int4*)sW1;
        int4* s2 = (int4*)sW2;
        for (int i = tid; i < 512; i += 128) s1[i] = g1[i];
        for (int i = tid; i < 2048; i += 128) s2[i] = g2[i];
      }
#endif
    }
#if defined(HAVE_TDM)
    if (wave == 0) __builtin_amdgcn_s_wait_tensorcnt(0);
#endif
    __syncthreads();  // staged weights visible to all waves

    if (net == 0) {  // pull T-net weights toward L2 while computing S-net
      const char* nx = (const char*)(W2s + (size_t)uT * 16384);
      for (int i = tid; i < 256; i += 128) __builtin_prefetch(nx + i * 128, 0, 0);
    }

    int col16 = lane & 15;
    int rbase = (lane >> 4) * 8;
    float m[8];

    // ---- layer 1: (16x32) @ (32x128), 8 column tiles ----
#pragma unroll
    for (int ct = 0; ct < 8; ++ct) {
      v16h a = load_a_frag(sEA[wave], lane, 0, 32);
      v16h bb = *(const v16h*)(sW1 + (ct * 32 + lane) * 16);
      v8f acc = {0.f, 0.f, 0.f, 0.f, 0.f, 0.f, 0.f, 0.f};
      acc = wmma_f16(a, bb, acc);
      int col = ct * 16 + col16;
      float bias1 = kb1[u * HDIM + col];
#pragma unroll
      for (int r = 0; r < 8; ++r) {
        float hv = acc[r] + bias1;
        hv = hv > 0.f ? hv : 0.f;
        sH1[wave][(rbase + r) * HDIM + col] = (_Float16)hv;
      }
    }

    // ---- layer 2: (16x128) @ (128x128), K chained, then max over rows ----
#pragma unroll
    for (int ct = 0; ct < 8; ++ct) {
      v8f acc = {0.f, 0.f, 0.f, 0.f, 0.f, 0.f, 0.f, 0.f};
#pragma unroll
      for (int chunk = 0; chunk < 4; ++chunk) {
        v16h a = load_a_frag(sH1[wave], lane, chunk * 32, HDIM);
        v16h bb = *(const v16h*)(sW2 + ((ct * 4 + chunk) * 32 + lane) * 16);
        acc = wmma_f16(a, bb, acc);
      }
      int col = ct * 16 + col16;
      float bias2 = kb2[u * HDIM + col];
      float rmax = -1e30f;
#pragma unroll
      for (int r = 0; r < 8; ++r) {
        float hv = acc[r] + bias2;
        hv = hv > 0.f ? hv : 0.f;
        rmax = fmaxf(rmax, hv);
      }
      rmax = fmaxf(rmax, __shfl_xor(rmax, 16, 32));  // both half-rows of col
      m[ct] = rmax;
    }

    // ---- projection 128 -> 13 via per-lane partials + shfl reduction ----
    const float* w3 = kW3 + (size_t)u * HDIM * C2;
    float partial[C2];
#pragma unroll
    for (int j = 0; j < C2; ++j) partial[j] = 0.f;
#pragma unroll
    for (int ct = 0; ct < 8; ++ct) {
      int colc = ct * 16 + col16;
      float mv = m[ct];
#pragma unroll
      for (int j = 0; j < C2; ++j) partial[j] += mv * w3[colc * C2 + j];
    }
#pragma unroll
    for (int j = 0; j < C2; ++j) {
      float pv = partial[j];
      pv += __shfl_xor(pv, 1, 32);
      pv += __shfl_xor(pv, 2, 32);
      pv += __shfl_xor(pv, 4, 32);
      pv += __shfl_xor(pv, 8, 32);
      outv[net][j] = pv + kb3[u * C2 + j];  // identical across all lanes
    }
  }

  // ---- coupling update (wave-local; outs are lane-uniform) ----
  float xb = (lane < C2) ? xrow[C1 + lane] : 0.f;
  float sel_s = 1.f, sel_t = 0.f, lsum = 0.f;
#pragma unroll
  for (int j = 0; j < C2; ++j) {
    float s = 1.f / (1.f + expf(-(outv[0][j] + 2.f)));
    float t = outv[1][j];
    lsum += logf(s);
    if (lane == j) { sel_s = s; sel_t = t; }
  }
  if (lane < C2)
    xrow[C1 + lane] = invmode ? (xb / sel_s - sel_t) : ((xb + sel_t) * sel_s);
  if (lane == 0 && !invmode) node_ldj[bn] = lsum;
}

// ---------------------------------------------------------------------------
// Small per-point / per-element kernels
// ---------------------------------------------------------------------------
__global__ void cond_kernel(const float* xyz, const float* sw1, const float* sb1,
                            const float* sw2, const float* sb2, float* c) {
  int p = blockIdx.x * 256 + threadIdx.x;
  if (p >= B_SZ * N_PTS) return;
  const float* xp = xyz + (size_t)p * PC;
  float h[32];
  for (int j = 0; j < 32; ++j) {
    float a = sb1[j] + xp[0] * sw1[j] + xp[1] * sw1[32 + j] + xp[2] * sw1[64 + j];
    h[j] = a > 0.f ? a : 0.f;
  }
  float* cp = c + (size_t)p * A_HID;
  for (int j = 0; j < A_HID; ++j) {
    float a = sb2[j];
    for (int i = 0; i < 32; ++i) a += h[i] * sw2[i * A_HID + j];
    cp[j] = a > 0.f ? a : 0.f;
  }
}

__global__ void neg_half_sq_kernel(const float* v, int C, float* pp) {
  int p = blockIdx.x * 256 + threadIdx.x;
  if (p >= B_SZ * N_PTS) return;
  float s = 0.f;
  for (int c = 0; c < C; ++c) { float x = v[(size_t)p * C + c]; s += x * x + LOG2PI; }
  pp[p] = -0.5f * s;
}

__global__ void reduce_b_kernel(const float* pp, float* acc, int add) {
  int b = threadIdx.x;
  if (b >= B_SZ) return;
  float s = 0.f;
  for (int n = 0; n < N_PTS; ++n) s += pp[b * N_PTS + n];
  acc[b] = (add ? acc[b] : 0.f) + s;
}

__global__ void reduce_step_kernel(const float* pp, const float* a_logs, int j, float* acc) {
  int b = threadIdx.x;
  if (b >= B_SZ) return;
  float s = 0.f;
  for (int n = 0; n < N_PTS; ++n) s += pp[b * N_PTS + n];
  float ls = 0.f;
  for (int c = 0; c < AUG; ++c) ls += a_logs[j * AUG + c];
  acc[b] += s + (float)N_PTS * ls;
}

__global__ void aug_step_kernel(float* y, const float* c,
                                const float* a_logs, const float* a_bias,
                                const float* aw1, const float* ab1,
                                const float* aw2, const float* ab2,
                                const float* aw3, const float* ab3,
                                int j, float* pldj) {
  int p = blockIdx.x * 128 + threadIdx.x;
  if (p >= B_SZ * N_PTS) return;
  float* yp = y + (size_t)p * AUG;
  float yv[AUG];
  for (int ch = 0; ch < AUG; ++ch)
    yv[ch] = yp[ch] * expf(a_logs[j * AUG + ch]) + a_bias[j * AUG + ch];
  float inp[AIN];
  for (int i = 0; i < A_C1; ++i) inp[i] = yv[i];
  for (int i = 0; i < A_HID; ++i) inp[A_C1 + i] = c[(size_t)p * A_HID + i];

  float outs[2][A_C1];
  for (int net = 0; net < 2; ++net) {
    int u = 2 * j + net;
    const float* W1 = aw1 + (size_t)u * AIN * A_HID;
    const float* W2 = aw2 + (size_t)u * A_HID * A_HID;
    const float* W3 = aw3 + (size_t)u * A_HID * A_C1;
    float h1[A_HID], h2[A_HID];
    for (int o = 0; o < A_HID; ++o) {
      float a = ab1[u * A_HID + o];
      for (int i = 0; i < AIN; ++i) a += inp[i] * W1[i * A_HID + o];
      h1[o] = a > 0.f ? a : 0.f;
    }
    for (int o = 0; o < A_HID; ++o) {
      float a = ab2[u * A_HID + o];
      for (int i = 0; i < A_HID; ++i) a += h1[i] * W2[i * A_HID + o];
      h2[o] = a > 0.f ? a : 0.f;
    }
    for (int o = 0; o < A_C1; ++o) {
      float a = ab3[u * A_C1 + o];
      for (int i = 0; i < A_HID; ++i) a += h2[i] * W3[i * A_C1 + o];
      outs[net][o] = a;
    }
  }
  float lsum = 0.f;
  for (int o = 0; o < A_C1; ++o) {
    float s = 1.f / (1.f + expf(-(outs[0][o] + 2.f)));
    float t = outs[1][o];
    yv[A_C1 + o] = (yv[A_C1 + o] + t) * s;
    lsum += logf(s);
  }
  for (int ch = 0; ch < AUG; ++ch) yp[ch] = yv[ch];
  pldj[p] = lsum;
}

__global__ void concat_kernel(const float* xyz, const float* y, float* x) {
  int t = blockIdx.x * 256 + threadIdx.x;
  if (t >= B_SZ * N_PTS * IDIM) return;
  int p = t / IDIM, ch = t % IDIM;
  x[t] = (ch < PC) ? xyz[(size_t)p * PC + ch] : y[(size_t)p * AUG + (ch - PC)];
}

__global__ void affine_kernel(float* x, const float* logs, const float* bias, int i, int inv) {
  int t = blockIdx.x * 256 + threadIdx.x;
  if (t >= B_SZ * N_PTS * IDIM) return;
  int ch = t % IDIM;
  float l = logs[i * IDIM + ch], bi = bias[i * IDIM + ch];
  x[t] = inv ? (x[t] - bi) * expf(-l) : (x[t] * expf(l) + bi);
}

__global__ void add_nlogsum_kernel(const float* logs, int i, float* acc) {
  if (blockIdx.x == 0 && threadIdx.x == 0) {
    float s = 0.f;
    for (int ch = 0; ch < IDIM; ++ch) s += logs[i * IDIM + ch];
    for (int b = 0; b < B_SZ; ++b) acc[b] += (float)N_PTS * s;
  }
}

__global__ void permute_kernel(const float* src, float* dst, const int* perms, int i, int inv) {
  int t = blockIdx.x * 256 + threadIdx.x;
  if (t >= B_SZ * N_PTS * IDIM) return;
  int p = t / IDIM, ch = t % IDIM;
  int pc = perms[i * IDIM + ch];
  if (!inv) dst[(size_t)p * IDIM + ch] = src[(size_t)p * IDIM + pc];
  else      dst[(size_t)p * IDIM + pc] = src[(size_t)p * IDIM + ch];
}

__global__ void zc_init_kernel(const float* z, float* zc) {
  int t = blockIdx.x * 256 + threadIdx.x;
  if (t >= B_SZ * N_PTS * IDIM) return;
  int ch = t % IDIM;
  zc[t] = (ch < IDIM - CUT) ? z[t] : 0.f;
}

__global__ void zero_kernel(float* a, int n) {
  int t = blockIdx.x * 256 + threadIdx.x;
  if (t < n) a[t] = 0.f;
}

__global__ void finalize_kernel(const float* zc, const float* acc, float* out) {
  int t = blockIdx.x * 256 + threadIdx.x;
  if (t < B_SZ * N_PTS * PC) {
    int p = t / PC, c = t % PC;
    out[t] = zc[(size_t)p * IDIM + c];
  } else if (t == B_SZ * N_PTS * PC) {
    // acc layout: [0..7] flow_ldj, [8..15] step_ldj, [16..23] logp_eps, [24..31] logpz
    float s = 0.f;
    for (int b = 0; b < B_SZ; ++b) {
      float aug_ldj = acc[16 + b] - acc[8 + b];
      s += acc[24 + b] + acc[b] - aug_ldj;
    }
    out[B_SZ * N_PTS * PC] = -s / (float)B_SZ;
  }
}

// ---------------------------------------------------------------------------
// Host orchestration
// ---------------------------------------------------------------------------
static void launch_knn(const float* pts, int stride, int C, int k, int* idx, hipStream_t s) {
  if (C == PC) {
    if (k == 8)       knn_kernel_t<8,  PC><<<B_SZ, N_PTS, 0, s>>>(pts, stride, idx);
    else if (k == 16) knn_kernel_t<16, PC><<<B_SZ, N_PTS, 0, s>>>(pts, stride, idx);
    else              knn_kernel_t<24, PC><<<B_SZ, N_PTS, 0, s>>>(pts, stride, idx);
  } else {
    knn_kernel_t<12, C1><<<B_SZ, N_PTS, 0, s>>>(pts, stride, idx);
  }
}

extern "C" void kernel_launch(void* const* d_in, const int* in_sizes, int n_in,
                              void* d_out, int out_size, void* d_ws, size_t ws_size,
                              hipStream_t stream) {
  const float* xyz    = (const float*)d_in[0];
  const float* eps    = (const float*)d_in[1];
  const float* sw1    = (const float*)d_in[2];
  const float* sb1    = (const float*)d_in[3];
  const float* sw2    = (const float*)d_in[4];
  const float* sb2    = (const float*)d_in[5];
  const float* a_logs = (const float*)d_in[6];
  const float* a_bias = (const float*)d_in[7];
  const float* aw1    = (const float*)d_in[8];
  const float* ab1    = (const float*)d_in[9];
  const float* aw2    = (const float*)d_in[10];
  const float* ab2    = (const float*)d_in[11];
  const float* aw3    = (const float*)d_in[12];
  const float* ab3    = (const float*)d_in[13];
  const float* f_logs = (const float*)d_in[14];
  const float* f_bias = (const float*)d_in[15];
  const float* kW1    = (const float*)d_in[16];
  const float* kb1    = (const float*)d_in[17];
  const float* kW2    = (const float*)d_in[18];
  const float* kb2    = (const float*)d_in[19];
  const float* kW3    = (const float*)d_in[20];
  const float* kb3    = (const float*)d_in[21];
  const int*   perms  = (const int*)d_in[22];
  float* out = (float*)d_out;

  const size_t BNX = (size_t)B_SZ * N_PTS * IDIM;   // 221184
  const size_t BN  = (size_t)B_SZ * N_PTS;          // 8192

  char* ws = (char*)d_ws;
  size_t off = 0;
  auto take = [&](size_t bytes) { size_t r = off; off += (bytes + 255) & ~(size_t)255; return r; };
  float*    Xa   = (float*)(ws + take(BNX * 4));
  float*    Xb   = (float*)(ws + take(BNX * 4));
  float*    Za   = (float*)(ws + take(BNX * 4));
  float*    Zb   = (float*)(ws + take(BNX * 4));
  float*    C64  = (float*)(ws + take(BN * A_HID * 4));
  float*    Y    = (float*)(ws + take(BN * AUG * 4));
  int*      IDX  = (int*)  (ws + take(BN * KMAX * 4));
  float*    NLDJ = (float*)(ws + take(BN * 4));
  float*    PP   = (float*)(ws + take(BN * 4));
  float*    ACC  = (float*)(ws + take(64 * 4));
  _Float16* W1s  = (_Float16*)(ws + take((size_t)NU * 4096 * 2));
  _Float16* W2s  = (_Float16*)(ws + take((size_t)NU * 16384 * 2));

  const int gX = (int)((BNX + 255) / 256);
  const int gP = (int)((BN + 255) / 256);
  const int gE = (int)(BN / NODES);     // edgeconv grid: 2048 blocks x 128 thr
  const int pre_ks[3] = {8, 16, 24};

  // -- weight prep + accumulator init --
  prep_w1_kernel<<<(NU * 4096 + 255) / 256, 256, 0, stream>>>(kW1, W1s);
  prep_w2_kernel<<<(NU * 16384 + 255) / 256, 256, 0, stream>>>(kW2, W2s);
  zero_kernel<<<1, 64, 0, stream>>>(ACC, 64);

  // -- conditioner + base log-prob of eps --
  cond_kernel<<<gP, 256, 0, stream>>>(xyz, sw1, sb1, sw2, sb2, C64);
  neg_half_sq_kernel<<<gP, 256, 0, stream>>>(eps, AUG, PP);
  reduce_b_kernel<<<1, B_SZ, 0, stream>>>(PP, ACC + 16, 1);

  // -- augmentation flow: y = eps through 3 coupling steps --
  hipMemcpyAsync(Y, eps, BN * AUG * sizeof(float), hipMemcpyDeviceToDevice, stream);
  for (int j = 0; j < 3; ++j) {
    aug_step_kernel<<<(int)((BN + 127) / 128), 128, 0, stream>>>(
        Y, C64, a_logs, a_bias, aw1, ab1, aw2, ab2, aw3, ab3, j, PP);
    reduce_step_kernel<<<1, B_SZ, 0, stream>>>(PP, a_logs, j, ACC + 8);
  }
  concat_kernel<<<gX, 256, 0, stream>>>(xyz, Y, Xa);

  // -- forward flow --
  float* X = Xa;
  float* XT = Xb;
  for (int i = 0; i < NFLOW; ++i) {
    int k = (i < 3) ? pre_ks[i] : 12;
    if (i < 3) launch_knn(xyz, PC, PC, k, IDX, stream);
    affine_kernel<<<gX, 256, 0, stream>>>(X, f_logs, f_bias, i, 0);
    add_nlogsum_kernel<<<1, 1, 0, stream>>>(f_logs, i, ACC);
    if (i >= 3) launch_knn(X, IDIM, C1, k, IDX, stream);
    edgeconv_kernel<<<gE, 128, 0, stream>>>(X, IDX, k, W1s, W2s, kb1, kb2, kW3, kb3,
                                            4 * i, 4 * i + 1, 0, NLDJ);
    reduce_b_kernel<<<1, B_SZ, 0, stream>>>(NLDJ, ACC, 1);
    permute_kernel<<<gX, 256, 0, stream>>>(X, XT, perms, i, 0);
    std::swap(X, XT);
    if (i >= 3) launch_knn(X, IDIM, C1, k, IDX, stream);
    edgeconv_kernel<<<gE, 128, 0, stream>>>(X, IDX, k, W1s, W2s, kb1, kb2, kW3, kb3,
                                            4 * i + 2, 4 * i + 3, 0, NLDJ);
    reduce_b_kernel<<<1, B_SZ, 0, stream>>>(NLDJ, ACC, 1);
  }

  // -- log p(z) + masked copy --
  neg_half_sq_kernel<<<gP, 256, 0, stream>>>(X, IDIM, PP);
  reduce_b_kernel<<<1, B_SZ, 0, stream>>>(PP, ACC + 24, 1);
  zc_init_kernel<<<gX, 256, 0, stream>>>(X, Za);

  // -- inverse flow on masked z --
  float* Z = Za;
  float* ZT = Zb;
  for (int i = NFLOW - 1; i >= 0; --i) {
    int k = (i < 3) ? pre_ks[i] : 12;
    if (i < 3) launch_knn(xyz, PC, PC, k, IDX, stream);
    else       launch_knn(Z, IDIM, C1, k, IDX, stream);
    edgeconv_kernel<<<gE, 128, 0, stream>>>(Z, IDX, k, W1s, W2s, kb1, kb2, kW3, kb3,
                                            4 * i + 2, 4 * i + 3, 1, NLDJ);
    permute_kernel<<<gX, 256, 0, stream>>>(Z, ZT, perms, i, 1);
    std::swap(Z, ZT);
    if (i >= 3) launch_knn(Z, IDIM, C1, k, IDX, stream);
    edgeconv_kernel<<<gE, 128, 0, stream>>>(Z, IDX, k, W1s, W2s, kb1, kb2, kW3, kb3,
                                            4 * i, 4 * i + 1, 1, NLDJ);
    affine_kernel<<<gX, 256, 0, stream>>>(Z, f_logs, f_bias, i, 1);
  }

  // -- outputs: clean_x then scalar ldj --
  finalize_kernel<<<(int)((BN * PC + 1 + 255) / 256), 256, 0, stream>>>(Z, ACC, out);
}
